// ScagRcsmFusion_7438883357133
// MI455X (gfx1250) — matrix operand order repaired and checked by reading, not compiled, not run on hardware
//
#include <hip/hip_runtime.h>
#include <hip/hip_bf16.h>
#include <math.h>

// ---------------------------------------------------------------------------
// ScagRcsmFusion for MI455X (gfx1250, wave32).
// Compute bound (~450 GFLOP vs ~100 MB traffic): all big GEMMs run on
// v_wmma_f32_16x16x32_f16. Tile staging uses the CDNA5 Tensor Data Mover
// (tensor_load_to_lds, TENSORcnt) with double-buffered LDS so the DMA for
// k-tile i+1 overlaps the 64 WMMAs of k-tile i.
// ---------------------------------------------------------------------------

typedef __attribute__((ext_vector_type(16))) _Float16 v16h;
typedef __attribute__((ext_vector_type(4)))  _Float16 h4v;
typedef __attribute__((ext_vector_type(2)))  _Float16 h2v;
typedef __attribute__((ext_vector_type(8)))  float    v8f;
typedef __attribute__((ext_vector_type(4)))  unsigned int u32x4;
typedef __attribute__((ext_vector_type(8)))  int      i32x8;
typedef __attribute__((ext_vector_type(4)))  int      i32x4;

#define BM 128
#define BN 128
#define BK 32
#define ROWS 34           // f32 dwords per LDS tile row (32 + 2 pad from TDM)

__device__ __forceinline__ float sigmoid_f(float x) { return 1.f / (1.f + expf(-x)); }
__device__ __forceinline__ float silu_f(float x)    { return x * sigmoid_f(x); }
__device__ __forceinline__ float softplus_f(float x){ return x > 20.f ? x : log1pf(expf(x)); }
__device__ __forceinline__ float gelu_f(float x)    { return 0.5f * x * (1.f + erff(x * 0.70710678118654752f)); }

// ---------------------------------------------------------------------------
// TDM 2-D tile load: rows x 32 f32, row stride (global) in f32 units.
// D# per CDNA5 ISA: group0 = {count, lds_addr, global_addr, type},
// group1 = {data_size=4B | pad_en | pad_interval=32DW | pad_amount=2DW, dims}.
// Generic LDS pointers carry the LDS offset in addr[31:0] (flat aperture), so
// truncation yields the D# lds_addr. This toolchain's builtin is the 6-arg
// form: (g0, g1, g2, g3, g4, cpol) — unused groups zero-filled.
// ---------------------------------------------------------------------------
__device__ __forceinline__ void tdm_load_tile(const float* gptr, const float* lds_ptr,
                                              unsigned rows, unsigned stride_f32)
{
    const unsigned long long ga = (unsigned long long)(uintptr_t)gptr;
    u32x4 g0;
    g0[0] = 1u;                                        // count=1 (valid), user mode
    g0[1] = (unsigned)(uintptr_t)lds_ptr;              // LDS byte address
    g0[2] = (unsigned)ga;                              // global addr [31:0]
    g0[3] = (unsigned)((ga >> 32) & 0x1FFFFFFu) | (2u << 30);  // [56:32] | type=2
    i32x8 g1;
    g1[0] = (int)((2u << 16) | (1u << 20) | (4u << 22) | (1u << 25));
    //            data_size=4B  pad_en     interval=32DW  amount=2DW
    g1[1] = (int)((32u & 0xFFFFu) << 16);                            // tensor_dim0 lo16
    g1[2] = (int)((rows & 0xFFFFu) << 16);                           // dim0 hi16 | tensor_dim1 lo16
    g1[3] = (int)(((rows >> 16) & 0xFFFFu) | (32u << 16));           // dim1 hi16 | tile_dim0=32
    g1[4] = (int)(rows & 0xFFFFu);                                   // tile_dim1 | tile_dim2=0
    g1[5] = (int)stride_f32;                                         // tensor_dim0_stride lo32
    g1[6] = 0;                                                       // stride hi16 | dim1_stride lo16
    g1[7] = 0;
    const i32x4 z4 = {0, 0, 0, 0};
    const i32x8 z8 = {0, 0, 0, 0, 0, 0, 0, 0};
    __builtin_amdgcn_tensor_load_to_lds(g0, g1, z4, z4, z8, 0);
}

// ---------------------------------------------------------------------------
// Fast GEMM: C[m,n] = EPI( sum_k A[m,k]*W[n,k] + bias[n] )  (x @ W.T)
// Requires M%128==0, N%128==0, K%32==0. 256 thr / 8 waves; wave tile 32x64
// (8 WMMAs per K-step). TDM double-buffered f32 tiles in LDS.
// ---------------------------------------------------------------------------
template <int EPI>
__global__ __launch_bounds__(256)
void wmma_gemm_tdm_kernel(const float* __restrict__ A, int lda,
                          const float* __restrict__ W, int ldw,
                          const float* __restrict__ bias,
                          float* __restrict__ C, int ldc,
                          int M, int N, int K)
{
    __shared__ float smem[4][BM * ROWS];   // [stage][A|B] : 2 stages x (A,B)

    const int tid   = threadIdx.x;
    const int lane  = tid & 31;
    const int wid   = tid >> 5;          // 0..7
    const int wm    = (wid >> 1) * 32;   // 4 waves along M
    const int wn    = (wid & 1) * 64;    // 2 waves along N
    const int bm    = blockIdx.y * BM;
    const int bn    = blockIdx.x * BN;
    const int m16   = lane & 15;
    const int khalf = lane >> 4;

    const int nk = K / BK;
    v8f acc[2][4] = {};

    if (wid == 0) {
        tdm_load_tile(A + (size_t)bm * lda, &smem[0][0], BM, (unsigned)lda);
        tdm_load_tile(W + (size_t)bn * ldw, &smem[1][0], BN, (unsigned)ldw);
    }

    for (int i = 0; i < nk; ++i) {
        const int cur = (i & 1) * 2;
        if (wid == 0) {
            if (i + 1 < nk) {
                const int nxt = ((i + 1) & 1) * 2;
                const int k1 = (i + 1) * BK;
                tdm_load_tile(A + (size_t)bm * lda + k1, &smem[nxt][0],     BM, (unsigned)lda);
                tdm_load_tile(W + (size_t)bn * ldw + k1, &smem[nxt + 1][0], BN, (unsigned)ldw);
                __builtin_amdgcn_s_wait_tensorcnt(2);   // stage i done, i+1 in flight
            } else {
                __builtin_amdgcn_s_wait_tensorcnt(0);
            }
        }
        __syncthreads();                                 // tile i visible to all waves

        const float* As = &smem[cur][0];
        const float* Bs = &smem[cur + 1][0];

        // ---- fragments per ISA 7.12.2 (wave32) layouts, f32->f16 on gather ----
        v16h afrag[2], bfrag[4];
#pragma unroll
        for (int r = 0; r < 2; ++r) {
            const int row = wm + r * 16 + m16;
#pragma unroll
            for (int v = 0; v < 8; ++v) {
                const int kk = (v < 4 ? 2 * v : 16 + 2 * (v - 4)) + khalf * 8;
                const float2 p = *(const float2*)&As[row * ROWS + kk];
                afrag[r][2 * v]     = (_Float16)p.x;
                afrag[r][2 * v + 1] = (_Float16)p.y;
            }
        }
#pragma unroll
        for (int c = 0; c < 4; ++c) {
            const int row = wn + c * 16 + m16;
#pragma unroll
            for (int v = 0; v < 8; ++v) {
                const int kk = (v < 4 ? 2 * v : 16 + 2 * (v - 4)) + khalf * 8;
                const float2 p = *(const float2*)&Bs[row * ROWS + kk];
                bfrag[c][2 * v]     = (_Float16)p.x;
                bfrag[c][2 * v + 1] = (_Float16)p.y;
            }
        }

#pragma unroll
        for (int r = 0; r < 2; ++r)
#pragma unroll
            for (int c = 0; c < 4; ++c)
                acc[r][c] = __builtin_amdgcn_wmma_f32_16x16x32_f16(
                    false, afrag[r], false, bfrag[c],
                    (short)0, acc[r][c], false, false);
        __syncthreads();                                 // readers done before DMA overwrite
    }

    // ---- epilogue (C/D layout: row = 8*(lane>>4)+v, col = lane&15) ----
#pragma unroll
    for (int r = 0; r < 2; ++r) {
#pragma unroll
        for (int c = 0; c < 4; ++c) {
            const int n = bn + wn + c * 16 + m16;
#pragma unroll
            for (int v = 0; v < 8; ++v) {
                const int m = bm + wm + r * 16 + khalf * 8 + v;
                float x = acc[r][c][v];
                if (bias) x += bias[n];
                if (EPI == 1) x = gelu_f(x);
                C[(size_t)m * ldc + n] = x;
            }
        }
    }
}

// ---------------------------------------------------------------------------
// Guarded GEMM for the two tiny ragged calls (N=80 / K=48) — synchronous
// f16 staging, per-element bounds checks.
// ---------------------------------------------------------------------------
#define LDSK (BK + 4)
template <int EPI>
__global__ __launch_bounds__(256)
void wmma_gemm_guard_kernel(const float* __restrict__ A, int lda,
                            const float* __restrict__ W, int ldw,
                            const float* __restrict__ bias,
                            float* __restrict__ C, int ldc,
                            int M, int N, int K)
{
    __shared__ _Float16 Ash[BM][LDSK];
    __shared__ _Float16 Bsh[BN][LDSK];

    const int tid   = threadIdx.x;
    const int lane  = tid & 31;
    const int wid   = tid >> 5;
    const int wm    = (wid >> 1) * 32;
    const int wn    = (wid & 1) * 64;
    const int bm    = blockIdx.y * BM;
    const int bn    = blockIdx.x * BN;
    const int m16   = lane & 15;
    const int khalf = lane >> 4;

    v8f acc[2][4] = {};

    for (int k0 = 0; k0 < K; k0 += BK) {
        for (int i = tid; i < BM * BK; i += 256) {
            const int r = i >> 5, c = i & 31;
            const int gk = k0 + c;
            const int gm = bm + r;
            Ash[r][c] = (_Float16)((gm < M && gk < K) ? A[(size_t)gm * lda + gk] : 0.f);
            const int gn = bn + r;
            Bsh[r][c] = (_Float16)((gn < N && gk < K) ? W[(size_t)gn * ldw + gk] : 0.f);
        }
        __syncthreads();

        v16h afrag[2], bfrag[4];
#pragma unroll
        for (int r = 0; r < 2; ++r) {
            const int row = wm + r * 16 + m16;
#pragma unroll
            for (int v = 0; v < 8; ++v) {
                const int kk = (v < 4 ? 2 * v : 16 + 2 * (v - 4)) + khalf * 8;
                h2v p = *(const h2v*)&Ash[row][kk];
                afrag[r][2 * v]     = p[0];
                afrag[r][2 * v + 1] = p[1];
            }
        }
#pragma unroll
        for (int c = 0; c < 4; ++c) {
            const int row = wn + c * 16 + m16;
#pragma unroll
            for (int v = 0; v < 8; ++v) {
                const int kk = (v < 4 ? 2 * v : 16 + 2 * (v - 4)) + khalf * 8;
                h2v p = *(const h2v*)&Bsh[row][kk];
                bfrag[c][2 * v]     = p[0];
                bfrag[c][2 * v + 1] = p[1];
            }
        }
#pragma unroll
        for (int r = 0; r < 2; ++r)
#pragma unroll
            for (int c = 0; c < 4; ++c)
                acc[r][c] = __builtin_amdgcn_wmma_f32_16x16x32_f16(
                    false, afrag[r], false, bfrag[c],
                    (short)0, acc[r][c], false, false);
        __syncthreads();
    }

#pragma unroll
    for (int r = 0; r < 2; ++r) {
#pragma unroll
        for (int c = 0; c < 4; ++c) {
            const int n = bn + wn + c * 16 + m16;
#pragma unroll
            for (int v = 0; v < 8; ++v) {
                const int m = bm + wm + r * 16 + khalf * 8 + v;
                if (m < M && n < N) {
                    float x = acc[r][c][v];
                    if (bias) x += bias[n];
                    if (EPI == 1) x = gelu_f(x);
                    C[(size_t)m * ldc + n] = x;
                }
            }
        }
    }
}

// ---------------------------------------------------------------------------
// Elementwise / reduction kernels
// ---------------------------------------------------------------------------

__global__ void c1_kernel(const float* __restrict__ cs, const float* __restrict__ w,
                          const float* __restrict__ b, float* __restrict__ y,
                          int B, int D)
{
    int i = blockIdx.x * blockDim.x + threadIdx.x;
    if (i >= B * D) return;
    y[i] = cs[i / D] * w[i % D] + b[i % D];
}

__global__ __launch_bounds__(256)
void row_ln_kernel(const float* __restrict__ X, const float* __restrict__ X2,
                   const float* __restrict__ g, const float* __restrict__ bta,
                   float* __restrict__ Y, int width, int do_relu)
{
    __shared__ float s_sum[256], s_sq[256];
    const int row = blockIdx.x;
    const float* x  = X  + (size_t)row * width;
    const float* x2 = X2 ? X2 + (size_t)row * width : nullptr;

    float xv[4];
    float sum = 0.f, sq = 0.f;
    int cnt = 0;
    for (int t = threadIdx.x; t < width; t += 256) {
        float v = x[t];
        if (x2) v += x2[t];
        xv[cnt++] = v;
        sum += v; sq += v * v;
    }
    s_sum[threadIdx.x] = sum; s_sq[threadIdx.x] = sq;
    __syncthreads();
    for (int off = 128; off > 0; off >>= 1) {
        if (threadIdx.x < off) {
            s_sum[threadIdx.x] += s_sum[threadIdx.x + off];
            s_sq[threadIdx.x]  += s_sq[threadIdx.x + off];
        }
        __syncthreads();
    }
    const float mu   = s_sum[0] / width;
    const float var  = s_sq[0] / width - mu * mu;
    const float rstd = rsqrtf(var + 1e-5f);

    cnt = 0;
    for (int t = threadIdx.x; t < width; t += 256) {
        float v = (xv[cnt++] - mu) * rstd;
        if (g) v = v * g[t] + bta[t];
        if (do_relu) v = fmaxf(v, 0.f);
        Y[(size_t)row * width + t] = v;
    }
}

__global__ void cw_kernel(const float* __restrict__ cs, const float* __restrict__ delta,
                          float* __restrict__ cw, int B, int D)
{
    int i = blockIdx.x * blockDim.x + threadIdx.x;
    if (i >= B * D) return;
    float l = 1.f - cs[i / D] + delta[i];
    l = fminf(fmaxf(l, -10.f), 10.f);
    cw[i] = sigmoid_f(l);
}

__global__ void concat2_kernel(const float* __restrict__ a, const float* __restrict__ b,
                               float* __restrict__ out, int B, int D)
{
    int i = blockIdx.x * blockDim.x + threadIdx.x;
    if (i >= B * 2 * D) return;
    int bi = i / (2 * D), d = i % (2 * D);
    out[i] = (d < D) ? a[(size_t)bi * D + d] : b[(size_t)bi * D + d - D];
}

__global__ void gate_stack_kernel(const float* __restrict__ mutp, const float* __restrict__ cw,
                                  const float* __restrict__ img_id, const float* __restrict__ img_attr,
                                  const float* __restrict__ txt_id, const float* __restrict__ txt_attr,
                                  float* __restrict__ s, int B, int D)
{
    int i = blockIdx.x * blockDim.x + threadIdx.x;
    if (i >= B * D) return;
    const int b = i / D, d = i % D;
    const float g0 = mutp[(size_t)b * 2 * D + d];
    const float g1 = mutp[(size_t)b * 2 * D + D + d];
    const float mx = fmaxf(g0, g1);
    const float e0 = expf(g0 - mx), e1 = expf(g1 - mx);
    const float inv = 1.f / (e0 + e1);
    const float c = cw[i];
    const float id = img_id[i], at = img_attr[i];
    const float gid = 0.7f * (id * (e0 * inv) * c) + 0.3f * id;
    const float gat = 0.7f * (at * (e1 * inv) * c) + 0.3f * at;
    const size_t base = ((size_t)b * 4) * D + d;
    s[base]         = gid;
    s[base + D]     = gat;
    s[base + 2 * D] = txt_id[i];
    s[base + 3 * D] = txt_attr[i];
}

__global__ void conv_silu_kernel(const float* __restrict__ xz, const float* __restrict__ conv_w,
                                 const float* __restrict__ conv_b, float* __restrict__ xc,
                                 int B, int rev)
{
    const int DIc = 1536;
    int i = blockIdx.x * blockDim.x + threadIdx.x;
    if (i >= B * 4 * DIc) return;
    const int d = i % DIc;
    const int t = (i / DIc) & 3;
    const int b = i / (DIc * 4);
    float acc = conv_b[d];
#pragma unroll
    for (int k = 0; k < 4; ++k) {
        const int tau = t - 3 + k;
        if (tau >= 0) {
            const int phys = rev ? 3 - tau : tau;
            acc += xz[((size_t)(b * 4 + phys)) * 3072 + d] * conv_w[d * 4 + k];
        }
    }
    xc[i] = silu_f(acc);
}

__global__ void scan_kernel(const float* __restrict__ dtpre, const float* __restrict__ xdbl,
                            const float* __restrict__ xc, const float* __restrict__ xz,
                            const float* __restrict__ A_log, const float* __restrict__ Dp,
                            float* __restrict__ ybuf, int B, int rev)
{
    const int DIc = 1536;
    int i = blockIdx.x * blockDim.x + threadIdx.x;
    if (i >= B * DIc) return;
    const int d = i % DIc;
    const int b = i / DIc;

    float a[16], h[16];
#pragma unroll
    for (int s = 0; s < 16; ++s) { a[s] = -expf(A_log[d * 16 + s]); h[s] = 0.f; }
    const float Dd = Dp[d];

    for (int t = 0; t < 4; ++t) {
        const size_t base = (size_t)(b * 4 + t);
        const float dtv  = softplus_f(dtpre[base * DIc + d]);
        const float xval = xc[base * DIc + d];
        const float* xd  = xdbl + base * 80;
        float y = 0.f;
#pragma unroll
        for (int s = 0; s < 16; ++s) {
            const float Bt = xd[48 + s];
            const float Ct = xd[64 + s];
            h[s] = expf(dtv * a[s]) * h[s] + dtv * xval * Bt;
            y += h[s] * Ct;
        }
        y += Dd * xval;
        const int phys = rev ? 3 - t : t;
        const float z = xz[((size_t)(b * 4 + phys)) * 3072 + 1536 + d];
        ybuf[base * DIc + d] = y * silu_f(z);
    }
}

__global__ void accum_kernel(float* __restrict__ sm, const float* __restrict__ yout,
                             int B, int D, int rev)
{
    int i = blockIdx.x * blockDim.x + threadIdx.x;
    if (i >= B * 4 * D) return;
    const int d = i % D;
    const int l = (i / D) & 3;
    const int b = i / (D * 4);
    const int t = rev ? 3 - l : l;
    sm[i] += yout[((size_t)(b * 4 + t)) * D + d];
}

__global__ void mean4_kernel(const float* __restrict__ so, float* __restrict__ ff, int B, int D)
{
    int i = blockIdx.x * blockDim.x + threadIdx.x;
    if (i >= B * D) return;
    const int b = i / D, d = i % D;
    const size_t base = ((size_t)b * 4) * D + d;
    ff[i] = 0.25f * (so[base] + so[base + D] + so[base + 2 * D] + so[base + 3 * D]);
}

__global__ __launch_bounds__(256)
void final_kernel(const float* __restrict__ outp, const float* __restrict__ res,
                  const float* __restrict__ g, const float* __restrict__ bta,
                  float* __restrict__ out, int D)
{
    __shared__ float s_sum[256], s_sq[256];
    const int b = blockIdx.x, t = threadIdx.x;
    const float v = (t < D) ? outp[(size_t)b * D + t] : 0.f;
    s_sum[t] = v; s_sq[t] = v * v;
    __syncthreads();
    for (int off = 128; off > 0; off >>= 1) {
        if (t < off) { s_sum[t] += s_sum[t + off]; s_sq[t] += s_sq[t + off]; }
        __syncthreads();
    }
    const float mu   = s_sum[0] / D;
    const float var  = s_sq[0] / D - mu * mu;
    const float rstd = rsqrtf(var + 1e-5f);
    if (t < D) {
        const float y = (v - mu) * rstd * g[t] + bta[t];
        out[(size_t)b * D + t] = 0.7f * y + 0.3f * res[(size_t)b * D + t];
    }
}

// ---------------------------------------------------------------------------
// Host orchestration
// ---------------------------------------------------------------------------
static inline void launch_gemm(const float* A, int lda, const float* W, int ldw,
                               const float* bias, float* C, int ldc,
                               int M, int N, int K, int epi, hipStream_t st)
{
    dim3 grid((N + BN - 1) / BN, (M + BM - 1) / BM);
    const bool guard = (M & (BM - 1)) || (N & (BN - 1)) || (K & (BK - 1));
    if (epi == 1) {
        if (guard) wmma_gemm_guard_kernel<1><<<grid, 256, 0, st>>>(A, lda, W, ldw, bias, C, ldc, M, N, K);
        else       wmma_gemm_tdm_kernel<1>  <<<grid, 256, 0, st>>>(A, lda, W, ldw, bias, C, ldc, M, N, K);
    } else {
        if (guard) wmma_gemm_guard_kernel<0><<<grid, 256, 0, st>>>(A, lda, W, ldw, bias, C, ldc, M, N, K);
        else       wmma_gemm_tdm_kernel<0>  <<<grid, 256, 0, st>>>(A, lda, W, ldw, bias, C, ldc, M, N, K);
    }
}

static inline dim3 eg(long long n) { return dim3((unsigned)((n + 255) / 256)); }

extern "C" void kernel_launch(void* const* d_in, const int* in_sizes, int n_in,
                              void* d_out, int out_size, void* d_ws, size_t ws_size,
                              hipStream_t stream)
{
    const int DM = 768, DI = 1536, L = 4, OUTD = 256;
    const int B = in_sizes[0] / DM;            // 4096
    const int BL = B * L;

#define IN(i) ((const float*)d_in[(i)])
    const float* img_id   = IN(0);
    const float* img_attr = IN(1);
    const float* txt_id   = IN(2);
    const float* txt_attr = IN(3);
    const float* csc      = IN(4);
    const float *c1_w = IN(5),  *c1_b = IN(6),  *ln1_g = IN(7),  *ln1_b = IN(8);
    const float *c2_w = IN(9),  *c2_b = IN(10), *ln2_g = IN(11), *ln2_b = IN(12);
    const float *c3_w = IN(13), *c3_b = IN(14);
    const float *mut_w = IN(15), *mut_b = IN(16);
    const float *ln_g = IN(17), *ln_b = IN(18);
    // mf: 19..27 (in_w, conv_w, conv_b, x_w, dt_w, dt_b, A_log, D, out_w); mb: 28..36
    const float *inj_w = IN(37), *inj_b = IN(38);
    const float *bn1_w = IN(39), *bn1_b = IN(40);
    const float *bn2_w = IN(41), *bn2_b = IN(42);
    const float *outp_w = IN(43), *outp_b = IN(44);
    const float *outn_g = IN(45), *outn_b = IN(46);
    const float *res_w = IN(47),  *res_b = IN(48);

    float* out0   = (float*)d_out;             // (B, 256)
    float* cw_out = out0 + (size_t)B * OUTD;   // (B, 768)

    float* w = (float*)d_ws;
    size_t off = 0;
    auto take = [&](size_t n) { float* p = w + off; off += n; return p; };
    float* xz    = take((size_t)BL * 3072);    // per-dir; reused as so1
    float* xc    = take((size_t)BL * DI);      // per-dir; reused as so2
    float* dtp   = take((size_t)BL * DI);
    float* ybuf  = take((size_t)BL * DI);
    float* s_buf = take((size_t)BL * DM);
    float* xn    = take((size_t)BL * DM);      // reused as fz
    float* sm    = take((size_t)BL * DM);
    float* yout  = take((size_t)BL * DM);      // reused as ss
    float* xdbl  = take((size_t)BL * 80);
    float* joint = take((size_t)B * 2 * DM);   // reused as ff
    float* mutp  = take((size_t)B * 2 * DM);   // reused as res
    float* h1    = take((size_t)B * 192);
    float* h2    = take((size_t)B * 384);
    float* delta = take((size_t)B * DM);       // reused as outp
    (void)ws_size;

    // ---- 1. conflict MLP -> cw ----
    c1_kernel<<<eg((long long)B * 192), 256, 0, stream>>>(csc, c1_w, c1_b, h1, B, 192);
    row_ln_kernel<<<B, 256, 0, stream>>>(h1, nullptr, ln1_g, ln1_b, h1, 192, 1);
    launch_gemm(h1, 192, c2_w, 192, c2_b, h2, 384, B, 384, 192, 0, stream);
    row_ln_kernel<<<B, 256, 0, stream>>>(h2, nullptr, ln2_g, ln2_b, h2, 384, 1);
    launch_gemm(h2, 384, c3_w, 384, c3_b, delta, DM, B, DM, 384, 0, stream);
    cw_kernel<<<eg((long long)B * DM), 256, 0, stream>>>(csc, delta, cw_out, B, DM);

    // ---- 2. mutual gates + stack + pre-LN ----
    concat2_kernel<<<eg((long long)B * 2 * DM), 256, 0, stream>>>(img_id, img_attr, joint, B, DM);
    launch_gemm(joint, 2 * DM, mut_w, 2 * DM, mut_b, mutp, 2 * DM, B, 2 * DM, 2 * DM, 0, stream);
    gate_stack_kernel<<<eg((long long)B * DM), 256, 0, stream>>>(
        mutp, cw_out, img_id, img_attr, txt_id, txt_attr, s_buf, B, DM);
    row_ln_kernel<<<BL, 256, 0, stream>>>(s_buf, nullptr, ln_g, ln_b, xn, DM, 0);
    (void)hipMemcpyAsync(sm, s_buf, (size_t)BL * DM * sizeof(float),
                         hipMemcpyDeviceToDevice, stream);      // sm = s

    // ---- 3. bidirectional Mamba (dir 0 = forward, 1 = flipped) ----
    for (int dir = 0; dir < 2; ++dir) {
        const int pb = dir ? 28 : 19;
        const float *in_w = IN(pb + 0), *conv_w = IN(pb + 1), *conv_b = IN(pb + 2);
        const float *x_w = IN(pb + 3),  *dt_w  = IN(pb + 4),  *dt_b  = IN(pb + 5);
        const float *A_log = IN(pb + 6), *Dp = IN(pb + 7),    *out_w = IN(pb + 8);

        launch_gemm(xn, DM, in_w, DM, nullptr, xz, 2 * DI, BL, 2 * DI, DM, 0, stream);
        conv_silu_kernel<<<eg((long long)BL * DI), 256, 0, stream>>>(xz, conv_w, conv_b, xc, B, dir);
        launch_gemm(xc, DI, x_w, DI, nullptr, xdbl, 80, BL, 80, DI, 0, stream);
        launch_gemm(xdbl, 80, dt_w, 48, dt_b, dtp, DI, BL, DI, 48, 0, stream);
        scan_kernel<<<eg((long long)B * DI), 256, 0, stream>>>(
            dtp, xdbl, xc, xz, A_log, Dp, ybuf, B, dir);
        launch_gemm(ybuf, DI, out_w, DI, nullptr, yout, DM, BL, DM, DI, 0, stream);
        accum_kernel<<<eg((long long)BL * DM), 256, 0, stream>>>(sm, yout, B, DM, dir);
    }

    // ---- 4. tail: inj, fuse-LN, FFN (GELU), mean, heads ----
    float* ss = yout;
    launch_gemm(s_buf, DM, inj_w, DM, inj_b, ss, DM, BL, DM, DM, 0, stream);
    float* fz = xn;
    row_ln_kernel<<<BL, 256, 0, stream>>>(sm, ss, nullptr, nullptr, fz, DM, 0);
    float* so1 = xz;
    launch_gemm(fz, DM, bn1_w, DM, bn1_b, so1, 4 * DM, BL, 4 * DM, DM, 1 /*GELU*/, stream);
    float* so2 = xc;
    launch_gemm(so1, 4 * DM, bn2_w, 4 * DM, bn2_b, so2, DM, BL, DM, 4 * DM, 0, stream);
    float* ff = joint;
    mean4_kernel<<<eg((long long)B * DM), 256, 0, stream>>>(so2, ff, B, DM);
    float* outp = delta;
    launch_gemm(ff, DM, outp_w, DM, outp_b, outp, OUTD, B, OUTD, DM, 0, stream);
    float* res = mutp;
    launch_gemm(img_id, DM, res_w, DM, res_b, res, OUTD, B, OUTD, DM, 0, stream);
    final_kernel<<<B, 256, 0, stream>>>(outp, res, outn_g, outn_b, out0, OUTD);

    (void)n_in; (void)out_size;
#undef IN
}
